// Head_31645319037451
// MI455X (gfx1250) — matrix operand order repaired and checked by reading, not compiled, not run on hardware
//
#include <hip/hip_runtime.h>
#include <hip/hip_bf16.h>

typedef __attribute__((ext_vector_type(16))) _Float16 v16h;
typedef __attribute__((ext_vector_type(8)))  _Float16 h8;
typedef __attribute__((ext_vector_type(8)))  float    v8f;
typedef __attribute__((ext_vector_type(4)))  float    f4;

static constexpr int B_  = 8;
static constexpr int T_  = 2048;
static constexpr int C_  = 1024;
static constexpr int H_  = 64;
static constexpr int BT_ = B_ * T_;   // 16384

// ---------------------------------------------------------------------------
// Kernel 1: transpose weights into f16 [3][H][C]; fold softmax scale into Wq.
// ---------------------------------------------------------------------------
__global__ __launch_bounds__(256) void prep_w(const float* __restrict__ Wk,
                                              const float* __restrict__ Wq,
                                              const float* __restrict__ Wv,
                                              _Float16* __restrict__ Wt) {
    int idx = blockIdx.x * 256 + threadIdx.x;       // over H*C
    if (idx >= H_ * C_) return;
    int h = idx / C_;
    int c = idx - h * C_;
    // order: [0]=Q (scaled), [1]=K, [2]=V
    Wt[0 * H_ * C_ + h * C_ + c] = (_Float16)(Wq[c * H_ + h] * 0.125f); // H^-0.5
    Wt[1 * H_ * C_ + h * C_ + c] = (_Float16)(Wk[c * H_ + h]);
    Wt[2 * H_ * C_ + h * C_ + c] = (_Float16)(Wv[c * H_ + h]);
}

// ---------------------------------------------------------------------------
// Kernel 2: fused Q/K/V projection.  One wave owns a 16-token tile and makes
// the full 16x64 output of all three matrices (12 accumulators) so x is read
// from HBM once.  The per-step 12KB weight slice [3][64][32] is staged into
// LDS cooperatively with double buffering; the 12-WMMA inner loop is software
// pipelined (fragment i+1 loads while WMMA i executes).
// Q,K stored row-major f16 [BT][H]; V stored transposed f16 [B][H][T].
// ---------------------------------------------------------------------------
__global__ __launch_bounds__(128) void proj_kernel(const float*    __restrict__ x,
                                                   const _Float16* __restrict__ Wt,
                                                   _Float16* __restrict__ Qh,
                                                   _Float16* __restrict__ Kh,
                                                   _Float16* __restrict__ Vt) {
    __shared__ _Float16 Wl[2][3 * H_ * 32];      // 2 x 12KB weight chunk

    const int tid  = threadIdx.x;
    const int wave = tid >> 5;
    const int lane = tid & 31;
    const int lr   = lane & 15;          // row/col within 16-tile
    const int grp  = lane >> 4;          // lane half (0/1)

    const int m0   = (blockIdx.x * 4 + wave) * 16;   // token tile base
    const int arow = m0 + lr;                        // A-matrix row = lane%16

    // cooperative copy mapping: 192 weight rows x 2 half-rows of 16 halves
    int crow[3], cpart[3];
    #pragma unroll
    for (int i = 0; i < 3; ++i) {
        int cid  = tid + 128 * i;        // 0..383
        crow[i]  = cid >> 1;             // 0..191  (= m*64 + h)
        cpart[i] = (cid & 1) * 16;       // 0 or 16
    }

    // ---- prologue: stage chunk k0=0, prefetch A(k0=0)
    h8 pf[6];
    #pragma unroll
    for (int i = 0; i < 3; ++i) {
        const _Float16* gp = Wt + (size_t)crow[i] * C_ + cpart[i];
        pf[2 * i]     = *(const h8*)(gp);
        pf[2 * i + 1] = *(const h8*)(gp + 8);
    }
    #pragma unroll
    for (int i = 0; i < 3; ++i) {
        _Float16* sp = &Wl[0][crow[i] * 32 + cpart[i]];
        *(h8*)(sp)     = pf[2 * i];
        *(h8*)(sp + 8) = pf[2 * i + 1];
    }
    __syncthreads();

    f4 af[4];
    {
        const float* xp = x + (size_t)arow * C_;
        af[0] = *(const f4*)(xp + 8 * grp);
        af[1] = *(const f4*)(xp + 8 * grp + 4);
        af[2] = *(const f4*)(xp + 16 + 8 * grp);
        af[3] = *(const f4*)(xp + 16 + 8 * grp + 4);
    }

    v8f acc[3][4];                       // [matrix][n-tile]
    #pragma unroll
    for (int m = 0; m < 3; ++m)
        #pragma unroll
        for (int n = 0; n < 4; ++n) acc[m][n] = (v8f){};

    const int fbase = lr * 32 + 8 * grp;     // per-lane B-frag LDS offset

    int p = 0;
    for (int k0 = 0; k0 < C_; k0 += 32, p ^= 1) {
        const bool more = (k0 + 32) < C_;

        // ---- issue next chunk's global loads (weights + x) before computing
        f4 an[4];
        if (more) {
            #pragma unroll
            for (int i = 0; i < 3; ++i) {
                const _Float16* gp = Wt + (size_t)crow[i] * C_ + (k0 + 32) + cpart[i];
                pf[2 * i]     = *(const h8*)(gp);
                pf[2 * i + 1] = *(const h8*)(gp + 8);
            }
            const float* xp = x + (size_t)arow * C_ + k0 + 32;
            an[0] = *(const f4*)(xp + 8 * grp);
            an[1] = *(const f4*)(xp + 8 * grp + 4);
            an[2] = *(const f4*)(xp + 16 + 8 * grp);
            an[3] = *(const f4*)(xp + 16 + 8 * grp + 4);
        }

        // ---- A fragment from prefetched x (fp32 -> f16)
        v16h a;
        #pragma unroll
        for (int j = 0; j < 4; ++j) {
            a[j]      = (_Float16)af[0][j];
            a[j + 4]  = (_Float16)af[1][j];
            a[j + 8]  = (_Float16)af[2][j];
            a[j + 12] = (_Float16)af[3][j];
        }

        // ---- 12 WMMAs fed from LDS, software pipelined (load i+1 || wmma i)
        const _Float16* lb = &Wl[p][fbase];      // frag idx f -> + f*512 halves
        v16h cur, nxt;
        {
            h8 b0 = *(const h8*)(lb);
            h8 b1 = *(const h8*)(lb + 16);
            #pragma unroll
            for (int e = 0; e < 8; ++e) { cur[e] = b0[e]; cur[e + 8] = b1[e]; }
        }
        #pragma unroll
        for (int f = 0; f < 12; ++f) {
            if (f < 11) {
                const _Float16* sp = lb + (f + 1) * (16 * 32);   // next frag
                h8 b0 = *(const h8*)(sp);
                h8 b1 = *(const h8*)(sp + 16);
                #pragma unroll
                for (int e = 0; e < 8; ++e) { nxt[e] = b0[e]; nxt[e + 8] = b1[e]; }
            }
            acc[f >> 2][f & 3] = __builtin_amdgcn_wmma_f32_16x16x32_f16(
                    false, a, false, cur, (short)0, acc[f >> 2][f & 3], false, false);
            if (f < 11) cur = nxt;
        }

        // ---- commit prefetched weights to the other LDS buffer
        if (more) {
            #pragma unroll
            for (int i = 0; i < 3; ++i) {
                _Float16* sp = &Wl[p ^ 1][crow[i] * 32 + cpart[i]];
                *(h8*)(sp)     = pf[2 * i];
                *(h8*)(sp + 8) = pf[2 * i + 1];
            }
            #pragma unroll
            for (int j = 0; j < 4; ++j) af[j] = an[j];
        }
        __syncthreads();
    }

    // ---- store: D layout -> lane col = n*16+lr, reg r -> row m0 + r + 8*grp
    #pragma unroll
    for (int n = 0; n < 4; ++n) {
        const int h = n * 16 + lr;
        #pragma unroll
        for (int r = 0; r < 8; ++r) {
            const int tok = m0 + r + 8 * grp;
            Qh[(size_t)tok * H_ + h] = (_Float16)acc[0][n][r];
            Kh[(size_t)tok * H_ + h] = (_Float16)acc[1][n][r];
            const int bb = tok >> 11;            // /T_
            const int tt = tok & (T_ - 1);
            Vt[((size_t)(bb * H_ + h)) * T_ + tt] = (_Float16)acc[2][n][r];
        }
    }
}

// ---------------------------------------------------------------------------
// Kernel 3: flash attention.  grid.x = T/16/4, grid.y = B.  4 waves/block,
// wave w handles query tile blockIdx.x*4+w.  32 keys per inner step.
// All K/V fragments for the step are loaded in one clause up front.
// ---------------------------------------------------------------------------
__global__ __launch_bounds__(128) void attn_kernel(const _Float16* __restrict__ Qh,
                                                   const _Float16* __restrict__ Kh,
                                                   const _Float16* __restrict__ Vt,
                                                   float* __restrict__ out) {
    __shared__ _Float16 Plds[4][16 * 32];    // per-wave P staging (D- -> A-layout)

    const int wave = threadIdx.x >> 5;
    const int lane = threadIdx.x & 31;
    const int lr   = lane & 15;
    const int grp  = lane >> 4;

    const int b  = blockIdx.y;
    const int qt = blockIdx.x * 4 + wave;    // query tile within batch
    const int q0 = qt * 16;

    // ---- Q A-fragments (h split in two K=32 pieces), loaded once
    const _Float16* qp = Qh + (size_t)(b * T_ + q0 + lr) * H_;
    v16h aq[2];
    #pragma unroll
    for (int j = 0; j < 2; ++j) {
        h8 x0 = *(const h8*)(qp + 32 * j + 8 * grp);
        h8 x1 = *(const h8*)(qp + 32 * j + 16 + 8 * grp);
        #pragma unroll
        for (int e = 0; e < 8; ++e) { aq[j][e] = x0[e]; aq[j][e + 8] = x1[e]; }
    }

    float mrow[8], lrow[8];
    #pragma unroll
    for (int r = 0; r < 8; ++r) { mrow[r] = -1e30f; lrow[r] = 0.0f; }
    v8f o[4];
    #pragma unroll
    for (int j = 0; j < 4; ++j) o[j] = (v8f){};

    _Float16* P = Plds[wave];
    const int nblk = q0 / 32 + 1;            // causal: keys <= q0+15

    for (int kb = 0; kb < nblk; ++kb) {
        const int s0 = kb * 32;

        // ---- bulk-load ALL K and V fragments for this 32-key block up front
        v16h kf[2][2];                       // [key-tile][h-half] B frags for QK^T
        #pragma unroll
        for (int st = 0; st < 2; ++st) {
            const _Float16* kp = Kh + (size_t)(b * T_ + s0 + st * 16 + lr) * H_;
            #pragma unroll
            for (int j = 0; j < 2; ++j) {
                h8 x0 = *(const h8*)(kp + 32 * j + 8 * grp);
                h8 x1 = *(const h8*)(kp + 32 * j + 16 + 8 * grp);
                #pragma unroll
                for (int e = 0; e < 8; ++e) { kf[st][j][e] = x0[e]; kf[st][j][e + 8] = x1[e]; }
            }
        }
        v16h vf[4];                          // B frags for P*V (V transposed)
        #pragma unroll
        for (int j = 0; j < 4; ++j) {
            const _Float16* vp = Vt + ((size_t)(b * H_ + j * 16 + lr)) * T_ + s0 + 8 * grp;
            h8 y0 = *(const h8*)(vp);
            h8 y1 = *(const h8*)(vp + 16);
            #pragma unroll
            for (int e = 0; e < 8; ++e) { vf[j][e] = y0[e]; vf[j][e + 8] = y1[e]; }
        }

        // ---- S = Q K^T : two 16-key tiles, each 2 WMMAs over h
        v8f s[2];
        #pragma unroll
        for (int st = 0; st < 2; ++st) {
            s[st] = (v8f){};
            #pragma unroll
            for (int j = 0; j < 2; ++j)
                s[st] = __builtin_amdgcn_wmma_f32_16x16x32_f16(false, aq[j], false, kf[st][j],
                                                               (short)0, s[st], false, false);
        }

        // ---- causal mask (branchless): elem (row q0+r+8*grp, col s0+16*st+lr)
        #pragma unroll
        for (int st = 0; st < 2; ++st)
            #pragma unroll
            for (int r = 0; r < 8; ++r)
                if (s0 + st * 16 + lr > q0 + r + 8 * grp) s[st][r] = -1e30f;

        // ---- online softmax; row lives in 16 lanes of one half -> xor 1/2/4/8
        float alpha[8];
        #pragma unroll
        for (int r = 0; r < 8; ++r) {
            float mx = fmaxf(s[0][r], s[1][r]);
            mx = fmaxf(mx, __shfl_xor(mx, 1, 32));
            mx = fmaxf(mx, __shfl_xor(mx, 2, 32));
            mx = fmaxf(mx, __shfl_xor(mx, 4, 32));
            mx = fmaxf(mx, __shfl_xor(mx, 8, 32));
            float mnew = fmaxf(mrow[r], mx);
            float p0 = __expf(s[0][r] - mnew);
            float p1 = __expf(s[1][r] - mnew);
            float rs = p0 + p1;
            rs += __shfl_xor(rs, 1, 32);
            rs += __shfl_xor(rs, 2, 32);
            rs += __shfl_xor(rs, 4, 32);
            rs += __shfl_xor(rs, 8, 32);
            alpha[r] = __expf(mrow[r] - mnew);
            lrow[r]  = lrow[r] * alpha[r] + rs;
            mrow[r]  = mnew;
            s[0][r] = p0; s[1][r] = p1;
        }
        #pragma unroll
        for (int j = 0; j < 4; ++j)
            #pragma unroll
            for (int r = 0; r < 8; ++r) o[j][r] *= alpha[r];

        // ---- P: D-layout -> LDS row-major [16][32] -> A-layout fragment
        #pragma unroll
        for (int r = 0; r < 8; ++r) {
            P[(r + 8 * grp) * 32 + lr]      = (_Float16)s[0][r];
            P[(r + 8 * grp) * 32 + 16 + lr] = (_Float16)s[1][r];
        }
        h8 p0 = *(const h8*)(P + lr * 32 + 8 * grp);        // K chunks per A layout
        h8 p1 = *(const h8*)(P + lr * 32 + 16 + 8 * grp);
        v16h ap;
        #pragma unroll
        for (int e = 0; e < 8; ++e) { ap[e] = p0[e]; ap[e + 8] = p1[e]; }

        // ---- O += P * V : 4 h-tiles from preloaded V fragments
        #pragma unroll
        for (int j = 0; j < 4; ++j)
            o[j] = __builtin_amdgcn_wmma_f32_16x16x32_f16(false, ap, false, vf[j],
                                                          (short)0, o[j], false, false);
    }

    // ---- epilogue: divide by row sum, write fp32 (B,T,H)
    #pragma unroll
    for (int j = 0; j < 4; ++j) {
        const int h = j * 16 + lr;
        #pragma unroll
        for (int r = 0; r < 8; ++r) {
            const int t = q0 + r + 8 * grp;
            out[((size_t)(b * T_ + t)) * H_ + h] = o[j][r] / lrow[r];
        }
    }
}

// ---------------------------------------------------------------------------
extern "C" void kernel_launch(void* const* d_in, const int* in_sizes, int n_in,
                              void* d_out, int out_size, void* d_ws, size_t ws_size,
                              hipStream_t stream) {
    const float* x  = (const float*)d_in[0];
    const float* Wk = (const float*)d_in[1];
    const float* Wq = (const float*)d_in[2];
    const float* Wv = (const float*)d_in[3];
    float* out = (float*)d_out;

    _Float16* ws16 = (_Float16*)d_ws;
    _Float16* Qh = ws16;                          // [BT][H]  (scale folded)
    _Float16* Kh = Qh + (size_t)BT_ * H_;         // [BT][H]
    _Float16* Vt = Kh + (size_t)BT_ * H_;         // [B][H][T]
    _Float16* Wt = Vt + (size_t)BT_ * H_;         // [3][H][C]

    prep_w<<<dim3((H_ * C_ + 255) / 256), dim3(256), 0, stream>>>(Wk, Wq, Wv, Wt);
    proj_kernel<<<dim3(BT_ / 64), dim3(128), 0, stream>>>(x, Wt, Qh, Kh, Vt);
    attn_kernel<<<dim3(T_ / 16 / 4, B_), dim3(128), 0, stream>>>(Qh, Kh, Vt, out);
}
